// DIM_12043088298039
// MI455X (gfx1250) — matrix operand order repaired and checked by reading, not compiled, not run on hardware
//
#include <hip/hip_runtime.h>
#include <hip/hip_bf16.h>

typedef _Float16 half_t;
typedef __attribute__((ext_vector_type(16))) _Float16 v16h;
typedef __attribute__((ext_vector_type(8)))  _Float16 v8h;
typedef __attribute__((ext_vector_type(8)))  float    v8f;

// ---------------- constants for this problem ----------------
#define B_  96
#define T_  48
#define LD_ 64
#define GD_ 64
#define MT_ 10
#define H_  128

// XOR-butterfly via DS_SWIZZLE_B32 group-of-32 mode:
// offset[15]=0, xor_mask=offset[14:10], or_mask=offset[9:5]=0, and_mask=offset[4:0]=0x1F
// lane' = (lane & 0x1F) ^ xor_mask  -- no index VALU, immediate pattern.
#define SWZ_XOR(x, mask_imm)                                                   \
    __int_as_float(__builtin_amdgcn_ds_swizzle(__float_as_int(x),              \
                                               ((mask_imm) << 10) | 0x1F))

// ---------------- prep kernel 1: Ag[i][h] = sum_k ls[i,k] * gw1[k,h], k = m*64+d
__global__ __launch_bounds__(256) void gather_gemm_g1(
    const float* __restrict__ locals_, const int* __restrict__ idx_t,
    const float* __restrict__ gw1, float* __restrict__ Ag)
{
    int gid = blockIdx.x * 256 + threadIdx.x;   // 96*128 threads
    int i = gid >> 7;
    int h = gid & 127;
    float acc = 0.0f;
    for (int m = 0; m < MT_; ++m) {
        int t = idx_t[i * MT_ + m];
        const float* lp = locals_ + ((size_t)i * T_ + t) * LD_;
        const float* wp = gw1 + (size_t)(m * LD_) * H_ + h;
#pragma unroll 8
        for (int d = 0; d < LD_; ++d)
            acc = fmaf(lp[d], wp[(size_t)d * H_], acc);
    }
    Ag[gid] = acc;
}

// ---------------- prep kernel 2: Y[r][h] = sum_{d<64} X[r*64+d] * W[d*128+h]
__global__ __launch_bounds__(256) void gemm64_128(
    const float* __restrict__ X, const float* __restrict__ W,
    float* __restrict__ Y)
{
    int gid = blockIdx.x * 256 + threadIdx.x;   // rows*128 threads (multiple of 256)
    int r = gid >> 7;
    int h = gid & 127;
    const float* xp = X + (size_t)r * 64;
    float acc = 0.0f;
#pragma unroll 8
    for (int d = 0; d < 64; ++d)
        acc = fmaf(xp[d], W[(size_t)d * H_ + h], acc);
    Y[gid] = acc;
}

// ---------------- prep kernel 3: WT[n*128+k] = (f16) W[k*128+n]
__global__ __launch_bounds__(256) void transpose_to_f16(
    const float* __restrict__ W, half_t* __restrict__ WT)
{
    int gid = blockIdx.x * 256 + threadIdx.x;   // 16384 threads
    int n = gid >> 7;
    int k = gid & 127;
    WT[gid] = (half_t)W[(size_t)k * H_ + n];
}

// ---------------- main fused layer2+layer3 WMMA kernel ----------------
// One block per aIdx; iterates the 6 j-blocks of 16 output columns.
// W2 B-fragments are held resident in registers across the whole loop
// (loaded once per block); h1 tile is double-buffered in LDS.
// 8 waves; wave w computes h2 columns [16w,16w+16) via 4x v_wmma_f32_16x16x32_f16
// per j-block, then fuses +b2, relu, *w3, ds_swizzle butterfly reduce,
// cross-wave LDS reduce, +b3.
__global__ __launch_bounds__(256) void mi_layer23_wmma(
    const float* __restrict__ Apre,   // [nA][128]
    const float* __restrict__ Bpre,   // [96][128]
    const float* __restrict__ b1,     // [128]
    const half_t* __restrict__ W2T,   // [128][128], n-major (transposed, f16)
    const float* __restrict__ b2,     // [128]
    const float* __restrict__ w3,     // [128]
    const float* __restrict__ b3,     // [1]
    float* __restrict__ out,
    int localMode)                    // 0: out[aIdx*96+j], 1: aIdx=(i*48+t), out[((t*96)+i)*96+j]
{
    __shared__ half_t h1s[2][16 * H_];    // double-buffered 16x128 f16 A tile
    __shared__ float  partial[8][16];     // per-wave layer-3 partials

    const int aIdx = blockIdx.x;

    const int tid  = threadIdx.x;
    const int wave = tid >> 5;
    const int lane = tid & 31;
    const int lmod = lane & 15;
    const int lhi  = lane >> 4;           // 0 or 1

    // ---- loop-invariant preloads ----
    // B fragments: 4 k-steps x v16h, resident in registers (32 VGPRs).
    // B k order used here (contiguous, same as A): lanes 0-15 hold K=0..15, lanes 16-31 K=16..31.
    const int n = wave * 16 + lmod;       // output column owned by this lane
    union frag { v16h v; v8h h[2]; };
    frag bf[4];
#pragma unroll
    for (int s = 0; s < 4; ++s) {
        const half_t* wp = W2T + (size_t)n * H_ + s * 32 + lhi * 16;
        bf[s].h[0] = *(const v8h*)(wp);
        bf[s].h[1] = *(const v8h*)(wp + 8);
    }
    const float b2n = b2[n];
    const float w3n = w3[n];
    const float b3v = b3[0];

    // Per-thread slice for h1-tile construction: row = tid>>4, cols c0..c0+7.
    const int row = tid >> 4;
    const int c0  = (tid & 15) * 8;
    float areg[8], b1reg[8];
    {
        const float* arow = Apre + (size_t)aIdx * H_;
#pragma unroll
        for (int u = 0; u < 8; ++u) {
            areg[u]  = arow[c0 + u];
            b1reg[u] = b1[c0 + u];
        }
    }

    // Output base row mapping for this aIdx.
    size_t obase;
    if (localMode) {
        int i = aIdx / T_;
        int t = aIdx % T_;
        obase = ((size_t)(t * B_ + i)) * B_;
    } else {
        obase = (size_t)aIdx * B_;
    }

    // ---- iterate the 6 column blocks of 16 ----
    for (int jb = 0; jb < 6; ++jb) {
        const int j0  = jb * 16;
        const int buf = jb & 1;
        half_t* tileW = &h1s[buf][0];

        // stage 1: build h1 tile in LDS (f16): relu(A + B[j0+row] + b1)
        {
            const float* brow = Bpre + (size_t)(j0 + row) * H_;
#pragma unroll
            for (int u = 0; u < 8; ++u) {
                float v = areg[u] + brow[c0 + u] + b1reg[u];
                tileW[row * H_ + c0 + u] = (half_t)fmaxf(v, 0.0f);
            }
        }
        __syncthreads();

        // stage 2: K=128 via 4 chained WMMAs against register-resident B
        const half_t* tileR = &h1s[buf][0];
        v8f acc = {};
#pragma unroll
        for (int s = 0; s < 4; ++s) {
            frag af;
            // A k order used here (contiguous, same as B): lanes 0-15 K=0..15, lanes 16-31 K=16..31
            const int ka = s * 32 + lhi * 16;
            af.h[0] = *(const v8h*)&tileR[lmod * H_ + ka];
            af.h[1] = *(const v8h*)&tileR[lmod * H_ + ka + 8];
            acc = __builtin_amdgcn_wmma_f32_16x16x32_f16(
                /*neg_a=*/false, af.v, /*neg_b=*/false, bf[s].v,
                /*c_mod=*/(short)0, acc, /*reuse_a=*/false, /*reuse_b=*/false);
        }

        // stage 3: fused layer 3: relu(h2 + b2) . w3, ds_swizzle XOR butterfly
        // over the 16 lanes sharing each row-half (masks <= 8 stay in-half).
        float vals[8];
#pragma unroll
        for (int r = 0; r < 8; ++r) {
            // lane holds D element (row = r + 8*lhi, col = n)
            float v = fmaxf(acc[r] + b2n, 0.0f) * w3n;
            v += SWZ_XOR(v, 1);
            v += SWZ_XOR(v, 2);
            v += SWZ_XOR(v, 4);
            v += SWZ_XOR(v, 8);
            vals[r] = v;
        }
        if (lmod == 0) {
#pragma unroll
            for (int r = 0; r < 8; ++r)
                partial[wave][lhi * 8 + r] = vals[r];
        }
        __syncthreads();

        // stage 4: deterministic cross-wave reduction + store 16 outputs
        if (tid < 16) {
            float s = b3v;
#pragma unroll
            for (int w = 0; w < 8; ++w) s += partial[w][tid];
            out[obase + j0 + tid] = s;
        }
        // No extra barrier needed: next iteration writes the other h1 buffer,
        // and `partial` is only rewritten after the next stage-1 barrier.
    }
}

extern "C" void kernel_launch(void* const* d_in, const int* in_sizes, int n_in,
                              void* d_out, int out_size, void* d_ws, size_t ws_size,
                              hipStream_t stream) {
    // inputs (setup_inputs order)
    const float* globals_ = (const float*)d_in[0];   // [96,64]
    const float* locals_  = (const float*)d_in[1];   // [96,48,64]
    const float* gw1 = (const float*)d_in[2];        // [704,128]
    const float* gb1 = (const float*)d_in[3];
    const float* gw2 = (const float*)d_in[4];        // [128,128]
    const float* gb2 = (const float*)d_in[5];
    const float* gw3 = (const float*)d_in[6];        // [128,1]
    const float* gb3 = (const float*)d_in[7];
    const float* lw1 = (const float*)d_in[8];        // [128,128]
    const float* lb1 = (const float*)d_in[9];
    const float* lw2 = (const float*)d_in[10];       // [128,128]
    const float* lb2 = (const float*)d_in[11];
    const float* lw3 = (const float*)d_in[12];       // [128,1]
    const float* lb3 = (const float*)d_in[13];
    const int*   idx_t = (const int*)d_in[14];       // [96,10]

    float* out = (float*)d_out;
    // output layout: globals_(6144) | locals_(294912) | globals_mi(9216) | locals_mi(442368)
    float* out_gmi = out + 6144 + 294912;
    float* out_lmi = out_gmi + 9216;

    // workspace layout (floats, then halves)
    float* Ag   = (float*)d_ws;                // 96*128
    float* Bg   = Ag + 96 * 128;               // 96*128
    float* Gloc = Bg + 96 * 128;               // 96*128
    float* Lloc = Gloc + 96 * 128;             // 4608*128
    half_t* gw2T = (half_t*)(Lloc + 4608 * 128); // 128*128 f16
    half_t* lw2T = gw2T + 128 * 128;             // 128*128 f16

    // passthrough outputs
    hipMemcpyAsync(out, globals_, (size_t)6144 * sizeof(float),
                   hipMemcpyDeviceToDevice, stream);
    hipMemcpyAsync(out + 6144, locals_, (size_t)294912 * sizeof(float),
                   hipMemcpyDeviceToDevice, stream);

    // prep
    gather_gemm_g1<<<(96 * 128) / 256, 256, 0, stream>>>(locals_, idx_t, gw1, Ag);
    gemm64_128<<<(96 * 128) / 256, 256, 0, stream>>>(globals_, gw1 + (size_t)640 * 128, Bg);
    gemm64_128<<<(4608 * 128) / 256, 256, 0, stream>>>(locals_, lw1, Lloc);       // locals_ as [4608,64]
    gemm64_128<<<(96 * 128) / 256, 256, 0, stream>>>(globals_, lw1 + (size_t)64 * 128, Gloc);
    transpose_to_f16<<<(128 * 128) / 256, 256, 0, stream>>>(gw2, gw2T);
    transpose_to_f16<<<(128 * 128) / 256, 256, 0, stream>>>(lw2, lw2T);

    // main fused WMMA passes: one block per aIdx, 6 j-blocks each,
    // W2 fragments register-resident across the whole block.
    mi_layer23_wmma<<<96, 256, 0, stream>>>(
        Ag, Bg, gb1, gw2T, gb2, gw3, gb3, out_gmi, /*localMode=*/0);
    mi_layer23_wmma<<<4608, 256, 0, stream>>>(
        Lloc, Gloc, lb1, lw2T, lb2, lw3, lb3, out_lmi, /*localMode=*/1);
}
